// AttnBlock_ln_66864050864855
// MI455X (gfx1250) — matrix operand order repaired and checked by reading, not compiled, not run on hardware
//
#include <hip/hip_runtime.h>
#include <math.h>

// ---------------------------------------------------------------------------
// Types for CDNA5 WMMA (gfx1250, wave32).
// ---------------------------------------------------------------------------
typedef __bf16 bf16_t;
typedef __attribute__((ext_vector_type(16))) __bf16 v16bf;
typedef __attribute__((ext_vector_type(8)))  float  v8f;
typedef unsigned int u32;
typedef unsigned long long u64;
typedef __attribute__((ext_vector_type(4))) u32 u32x4;
typedef __attribute__((ext_vector_type(4))) int i32x4;
typedef __attribute__((ext_vector_type(8))) int i32x8;

struct FragBits { u32x4 lo, hi; };   // 32 bytes == v16bf

#ifndef __has_builtin
#define __has_builtin(x) 0
#endif
#if __has_builtin(__builtin_amdgcn_tensor_load_to_lds) && \
    __has_builtin(__builtin_amdgcn_s_wait_tensorcnt)
#define HAVE_TDM 1
#else
#define HAVE_TDM 0
#endif

static __device__ inline v8f vzero8() {
    v8f z = {0.f,0.f,0.f,0.f,0.f,0.f,0.f,0.f};
    return z;
}

// float -> bf16 round-to-nearest-even via bit manipulation (no __bf16 arith).
static __device__ inline bf16_t tobf(float x) {
    u32 u = __builtin_bit_cast(u32, x);
    u32 r = (u + 0x7fffu + ((u >> 16) & 1u)) >> 16;
    unsigned short h = (unsigned short)r;
    return __builtin_bit_cast(bf16_t, h);
}

static __device__ inline v16bf ld_frag2(const bf16_t* p0, const bf16_t* p1) {
    FragBits f;
    f.lo = *reinterpret_cast<const u32x4*>(p0);
    f.hi = *reinterpret_cast<const u32x4*>(p1);
    return __builtin_bit_cast(v16bf, f);
}

// A-matrix fragment (16x32 bf16, row-major [16, ld], K-contiguous):
// lanes 0-15: row=lane, K chunks {k0..k0+7, k0+16..k0+23}
// lanes 16-31: row=lane-16, K chunks {k0+8..k0+15, k0+24..k0+31}
static __device__ inline v16bf ld_fragA(const bf16_t* base, int ld, int lane, int k0) {
    const bf16_t* p = base + (size_t)(lane & 15) * ld + k0 + ((lane >> 4) << 3);
    return ld_frag2(p, p + 16);
}

// B-matrix fragment loaded as rows of B^T (row-major [16, ld]):
// lane n = lane&15 holds K chunks {k0+8h..k0+8h+7, k0+16+8h..k0+16+8h+7}, h = lane>>4.
static __device__ inline v16bf ld_fragB(const bf16_t* base, int ld, int lane, int k0) {
    const bf16_t* p = base + (size_t)(lane & 15) * ld + k0 + ((lane >> 4) << 3);
    return ld_frag2(p, p + 16);
}

static __device__ inline v8f wmma_bf16(v16bf a, v16bf b, v8f c) {
    return __builtin_amdgcn_wmma_f32_16x16x32_bf16(
        /*neg_a=*/false, a, /*neg_b=*/false, b,
        /*c_mod=*/(short)0, c, /*reuse_a=*/false, /*reuse_b=*/false);
}

#if HAVE_TDM
// Low 32 bits of a generic shared-memory pointer == LDS byte address
// (flat aperture scheme).
static __device__ inline u32 lds_lo(const void* p) {
    return (u32)(size_t)p;
}

// Issue a 2-D TDM tile load (HBM -> LDS), bf16 elements (data_size = 2B).
// Descriptor packing per CDNA5 ISA:
//  group0: [1:0]=count=1, [63:32]=lds_addr, [120:64]=global_addr, [127:126]=type(2)
//  group1: [17:16]=data_size(1 => 2B), tensor_dim0/1, tile_dim0/1 (tile_dim2=0),
//          tensor_dim0_stride (elements). groups 2/3 unused (2-D tensor).
static __device__ inline void tdm_load_2d(u32 lds_addr, const void* gaddr,
                                          u32 tensor_d0, u32 tensor_d1,
                                          u32 tile_d0, u32 tile_d1, u32 stride0) {
    u64 ga = (u64)(size_t)gaddr;
    u32x4 g0;
    g0.x = 0x1u;                                       // count = 1 (valid D#)
    g0.y = lds_addr;                                   // LDS byte address
    g0.z = (u32)ga;                                    // global_addr[31:0]
    g0.w = (u32)((ga >> 32) & 0x1FFFFFFu) | (2u << 30); // addr[56:32] | type=2
    i32x8 g1;
    g1[0] = (int)(1u << 16);                           // data_size = 2 bytes
    g1[1] = (int)((tensor_d0 & 0xFFFFu) << 16);        // tensor_dim0[15:0]
    g1[2] = (int)((tensor_d0 >> 16) | ((tensor_d1 & 0xFFFFu) << 16));
    g1[3] = (int)((tensor_d1 >> 16) | (tile_d0 << 16)); // tile_dim0
    g1[4] = (int)(tile_d1 & 0xFFFFu);                  // tile_dim1 (tile_dim2=0)
    g1[5] = (int)stride0;                              // tensor_dim0_stride[31:0]
    g1[6] = 0;                                         // stride hi, dim1_stride lo
    g1[7] = 0;
    i32x4 gz = {0, 0, 0, 0};
#if defined(__clang_major__) && (__clang_major__ >= 23)
    i32x8 gz8 = {0, 0, 0, 0, 0, 0, 0, 0};
    __builtin_amdgcn_tensor_load_to_lds(g0, g1, gz, gz, gz8, 0);
#else
    __builtin_amdgcn_tensor_load_to_lds(g0, g1, gz, gz, 0);
#endif
}
#endif  // HAVE_TDM

// ---------------------------------------------------------------------------
// Elementwise pack kernels
// ---------------------------------------------------------------------------
__global__ void pack_bf16_kernel(const float* __restrict__ src,
                                 bf16_t* __restrict__ dst, int n) {
    int i = blockIdx.x * blockDim.x + threadIdx.x;
    if (i < n) dst[i] = tobf(src[i]);
}

// src f32 [C, N] -> dst bf16 [N, ld] at column offset coloff (transpose+cast)
__global__ void packT_kernel(const float* __restrict__ src,
                             bf16_t* __restrict__ dst,
                             int C, int N, int ld, int coloff) {
    int i = blockIdx.x * blockDim.x + threadIdx.x;
    if (i >= C * N) return;
    int c = i / N, n = i - c * N;
    dst[(size_t)n * ld + coloff + c] = tobf(src[i]);
}

// Q,K,V f32 [256,2048] -> per-head: qh/kh bf16 [4][2048][64], vh bf16 [4][64][2048]
// split_heads channel mapping: channel = d*4 + h
__global__ void head_pack_kernel(const float* __restrict__ Qf,
                                 const float* __restrict__ Kf,
                                 const float* __restrict__ Vf,
                                 bf16_t* __restrict__ qh,
                                 bf16_t* __restrict__ kh,
                                 bf16_t* __restrict__ vh) {
    int i = blockIdx.x * blockDim.x + threadIdx.x; // over 4*64*2048
    if (i >= 4 * 64 * 2048) return;
    int h   = i >> 17;
    int rem = i & ((1 << 17) - 1);
    int d   = rem >> 11;
    int n   = rem & 2047;
    size_t srci = (size_t)(d * 4 + h) * 2048 + n;
    qh[((size_t)(h * 2048 + n)) * 64 + d]  = tobf(Qf[srci]);
    kh[((size_t)(h * 2048 + n)) * 64 + d]  = tobf(Kf[srci]);
    vh[((size_t)(h * 64 + d)) * 2048 + n]  = tobf(Vf[srci]);
}

// ---------------------------------------------------------------------------
// Generic bf16 WMMA GEMM:  C[M,N] = A[M,K] x B[K,N] (+bias[M]) (+resid[M,N])
// A row-major bf16; Bt = B^T row-major bf16 [N,K].
// Block = 128 threads (4 waves): 32x64 output tile; wave w owns 16 columns.
// ---------------------------------------------------------------------------
__global__ __launch_bounds__(128)
void gemm_kernel(const bf16_t* __restrict__ A, const bf16_t* __restrict__ Bt,
                 const float* __restrict__ bias, const float* __restrict__ resid,
                 float* __restrict__ C, int M, int N, int K) {
    int lane = threadIdx.x & 31;
    int w    = threadIdx.x >> 5;
    int hl   = lane >> 4;
    int ln15 = lane & 15;
    int n0   = blockIdx.x * 64;
    int m0   = blockIdx.y * 32;

    v8f acc0 = vzero8();
    v8f acc1 = vzero8();

    const bf16_t* a0base = A + (size_t)m0 * K;
    const bf16_t* a1base = A + (size_t)(m0 + 16) * K;
    const bf16_t* bbase  = Bt + (size_t)(n0 + w * 16) * K;

    for (int kk = 0; kk < K; kk += 32) {
        v16bf bf  = ld_fragB(bbase,  K, lane, kk);
        v16bf af0 = ld_fragA(a0base, K, lane, kk);
        acc0 = wmma_bf16(af0, bf, acc0);
        v16bf af1 = ld_fragA(a1base, K, lane, kk);
        acc1 = wmma_bf16(af1, bf, acc1);
    }

    int col = n0 + w * 16 + ln15;
    for (int r = 0; r < 8; ++r) {
        int row0 = m0 + r + 8 * hl;
        int row1 = row0 + 16;
        float v0 = acc0[r];
        float v1 = acc1[r];
        if (bias) { v0 += bias[row0]; v1 += bias[row1]; }
        size_t i0 = (size_t)row0 * N + col;
        size_t i1 = (size_t)row1 * N + col;
        if (resid) { v0 += resid[i0]; v1 += resid[i1]; }
        C[i0] = v0;
        C[i1] = v1;
    }
}

// ---------------------------------------------------------------------------
// Fused flash-style attention for one head (grid.y = head, grid.x = n-tile).
// qh,kh: bf16 [H][2048][64]; vh: bf16 [H][64][2048].
// Output written as [256][2048] f32 with channel = d*4 + h (split_heads merge).
//
// Block = 128 threads (4 waves); 32 query rows; key tiles of 64.
// Q fragments are loop-invariant -> preloaded into VGPRs.
// K/V tiles are staged into double-buffered LDS by the Tensor Data Mover
// (tensor_load_to_lds, wave 0 issues, s_wait_tensorcnt + barrier publishes),
// overlapping the next tile's DMA with this tile's WMMA + softmax.
// S/P tiles live only in LDS; scores never touch HBM.
// ---------------------------------------------------------------------------
__global__ __launch_bounds__(128)
void attn_kernel(const bf16_t* __restrict__ qh, const bf16_t* __restrict__ kh,
                 const bf16_t* __restrict__ vh, float* __restrict__ outp) {
    constexpr int Nn = 2048, HDm = 64;
    const float scale = 0.0625f;  // 1/sqrt(d_model=256)

    __shared__ __align__(16) bf16_t Kt[2][64][64];   // key tile  [m-local][d]
    __shared__ __align__(16) bf16_t Vt[2][64][64];   // val tile  [d][m-local]
    __shared__ __align__(16) float  Sm[32][64];
    __shared__ __align__(16) bf16_t Pm[32][64];
    __shared__ float alphaS[32];
    __shared__ float lsumS[32];

    int lane = threadIdx.x & 31;
    int w    = threadIdx.x >> 5;
    int hl   = lane >> 4;
    int ln15 = lane & 15;
    int h    = blockIdx.y;
    int n0   = blockIdx.x * 32;

    const bf16_t* q  = qh + (size_t)h * Nn * HDm;
    const bf16_t* k  = kh + (size_t)h * Nn * HDm;
    const bf16_t* vp = vh + (size_t)h * HDm * Nn;

    // Loop-invariant Q fragments in registers (4 x v16bf = 32 VGPRs).
    v16bf qf[2][2];
    for (int i = 0; i < 2; ++i) {
        const bf16_t* qb = q + (size_t)(n0 + i * 16) * HDm;
        qf[i][0] = ld_fragA(qb, HDm, lane, 0);
        qf[i][1] = ld_fragA(qb, HDm, lane, 32);
    }

    v8f acc0 = vzero8();
    v8f acc1 = vzero8();
    float m_i = -3.0e38f, l_i = 0.f;  // valid in threads < 32

#if HAVE_TDM
    if (w == 0) {   // one wave issues the DMA; TDM ignores EXEC
        tdm_load_2d(lds_lo(&Kt[0][0][0]), k,  HDm, Nn, HDm, 64, HDm);
        tdm_load_2d(lds_lo(&Vt[0][0][0]), vp, Nn, HDm, 64, HDm, Nn);
    }
#endif

    int buf = 0;
    for (int m0 = 0; m0 < Nn; m0 += 64) {
#if HAVE_TDM
        if (w == 0) __builtin_amdgcn_s_wait_tensorcnt((short)0);
        __syncthreads();   // tile[buf] now visible to all waves
        if (w == 0 && (m0 + 64) < Nn) {   // overlap next tile's DMA with compute
            tdm_load_2d(lds_lo(&Kt[buf ^ 1][0][0]),
                        k + (size_t)(m0 + 64) * HDm, HDm, Nn, HDm, 64, HDm);
            tdm_load_2d(lds_lo(&Vt[buf ^ 1][0][0]),
                        vp + (m0 + 64), Nn, HDm, 64, HDm, Nn);
        }
#else
        // Fallback: cooperative global->LDS staging.
        for (int idx = threadIdx.x; idx < 64 * 32; idx += 128) {
            int r = idx >> 5, c4 = idx & 31;
            reinterpret_cast<u32*>(&Kt[buf][r][0])[c4] =
                reinterpret_cast<const u32*>(k + (size_t)(m0 + r) * HDm)[c4];
            reinterpret_cast<u32*>(&Vt[buf][r][0])[c4] =
                reinterpret_cast<const u32*>(vp + (size_t)r * Nn + m0)[c4];
        }
        __syncthreads();
#endif

        // ---- S tile (32x64) from LDS K tile: wave w owns 16 m-columns ----
        v16bf kb0 = ld_fragB(&Kt[buf][w * 16][0], 64, lane, 0);
        v16bf kb1 = ld_fragB(&Kt[buf][w * 16][0], 64, lane, 32);
        for (int i = 0; i < 2; ++i) {
            v8f c = vzero8();
            c = wmma_bf16(qf[i][0], kb0, c);
            c = wmma_bf16(qf[i][1], kb1, c);
            for (int r = 0; r < 8; ++r)
                Sm[i * 16 + r + 8 * hl][w * 16 + ln15] = c[r] * scale;
        }
        __syncthreads();

        // ---- online softmax (one thread per row) ----
        if (threadIdx.x < 32) {
            int rr = threadIdx.x;
            float mt = -3.0e38f;
            for (int cc = 0; cc < 64; ++cc) mt = fmaxf(mt, Sm[rr][cc]);
            float mnew = fmaxf(m_i, mt);
            float al   = __expf(m_i - mnew);
            float ssum = 0.f;
            for (int cc = 0; cc < 64; ++cc) {
                float pv = __expf(Sm[rr][cc] - mnew);
                ssum += pv;
                Pm[rr][cc] = tobf(pv);
            }
            l_i = l_i * al + ssum;
            m_i = mnew;
            alphaS[rr] = al;
        }
        __syncthreads();

        // ---- rescale accumulators then O += P x V (wave w owns 16 d-cols) --
        for (int r = 0; r < 8; ++r) {
            acc0[r] *= alphaS[r + 8 * hl];
            acc1[r] *= alphaS[16 + r + 8 * hl];
        }
        v16bf vb0 = ld_fragB(&Vt[buf][w * 16][0], 64, lane, 0);
        v16bf vb1 = ld_fragB(&Vt[buf][w * 16][0], 64, lane, 32);
        v16bf p00 = ld_fragA(&Pm[0][0], 64, lane, 0);
        acc0 = wmma_bf16(p00, vb0, acc0);
        v16bf p01 = ld_fragA(&Pm[0][0], 64, lane, 32);
        acc0 = wmma_bf16(p01, vb1, acc0);
        v16bf p10 = ld_fragA(&Pm[16][0], 64, lane, 0);
        acc1 = wmma_bf16(p10, vb0, acc1);
        v16bf p11 = ld_fragA(&Pm[16][0], 64, lane, 32);
        acc1 = wmma_bf16(p11, vb1, acc1);
        __syncthreads();
        buf ^= 1;
    }

    if (threadIdx.x < 32) lsumS[threadIdx.x] = l_i;
    __syncthreads();

    int dcol = w * 16 + ln15;
    for (int r = 0; r < 8; ++r) {
        int row0 = r + 8 * hl;
        float o0 = acc0[r] / lsumS[row0];
        float o1 = acc1[r] / lsumS[16 + row0];
        size_t ch = (size_t)(dcol * 4 + h) * Nn;   // merged channel = d*4 + h
        outp[ch + n0 + row0]       = o0;
        outp[ch + n0 + 16 + row0]  = o1;
    }
}

// ---------------------------------------------------------------------------
// LayerNorm over channel (C=512) + exact GELU; output transposed bf16 [N, 512]
// ready for the W2 GEMM.
// ---------------------------------------------------------------------------
__global__ __launch_bounds__(256)
void ln_gelu_kernel(const float* __restrict__ h1, const float* __restrict__ g,
                    const float* __restrict__ bb, bf16_t* __restrict__ hT) {
    constexpr int C = 512, Nn = 2048;
    __shared__ float r1[256];
    __shared__ float r2[256];
    int n = blockIdx.x, t = threadIdx.x;
    float s1 = 0.f, s2 = 0.f;
    for (int c = t; c < C; c += 256) {
        float x = h1[(size_t)c * Nn + n];
        s1 += x; s2 += x * x;
    }
    r1[t] = s1; r2[t] = s2;
    __syncthreads();
    for (int off = 128; off > 0; off >>= 1) {
        if (t < off) { r1[t] += r1[t + off]; r2[t] += r2[t + off]; }
        __syncthreads();
    }
    float mean = r1[0] * (1.f / C);
    float var  = r2[0] * (1.f / C) - mean * mean;
    float rstd = rsqrtf(var + 1e-5f);
    for (int c = t; c < C; c += 256) {
        float x  = h1[(size_t)c * Nn + n];
        float xn = (x - mean) * rstd * g[c] + bb[c];
        float ge = 0.5f * xn * (1.0f + erff(xn * 0.70710678118654752f));
        hT[(size_t)n * C + c] = tobf(ge);
    }
}

// ---------------------------------------------------------------------------
// Host-side orchestration
// ---------------------------------------------------------------------------
template <typename T>
static T* carve(char*& p, size_t n) {
    T* r = (T*)p;
    p += (n * sizeof(T) + 255) & ~(size_t)255;
    return r;
}

extern "C" void kernel_launch(void* const* d_in, const int* in_sizes, int n_in,
                              void* d_out, int out_size, void* d_ws, size_t ws_size,
                              hipStream_t stream) {
    (void)in_sizes; (void)n_in; (void)out_size; (void)ws_size;
    constexpr int Dd = 256, Nn = 2048, D2 = 512, D3 = 768;
    constexpr size_t DN = (size_t)Dd * Nn;

    const float* desc[2] = {(const float*)d_in[0], (const float*)d_in[1]};
    const float* Wq = (const float*)d_in[2];  const float* bq = (const float*)d_in[3];
    const float* Wk = (const float*)d_in[4];  const float* bk = (const float*)d_in[5];
    const float* Wv = (const float*)d_in[6];  const float* bv = (const float*)d_in[7];
    const float* Wm = (const float*)d_in[8];  const float* bm = (const float*)d_in[9];
    const float* W1 = (const float*)d_in[10]; const float* b1 = (const float*)d_in[11];
    const float* lng = (const float*)d_in[12]; const float* lnb = (const float*)d_in[13];
    const float* W2 = (const float*)d_in[14]; const float* b2 = (const float*)d_in[15];
    float* out = (float*)d_out;

    char* p = (char*)d_ws;
    bf16_t* Wqb = carve<bf16_t>(p, (size_t)Dd * Dd);
    bf16_t* Wkb = carve<bf16_t>(p, (size_t)Dd * Dd);
    bf16_t* Wvb = carve<bf16_t>(p, (size_t)Dd * Dd);
    bf16_t* Wmb = carve<bf16_t>(p, (size_t)Dd * Dd);
    bf16_t* W1b = carve<bf16_t>(p, (size_t)D2 * D3);
    bf16_t* W2b = carve<bf16_t>(p, (size_t)Dd * D2);

    bf16_t* descT[4]; float* Qf[4]; float* Kf[4]; float* Vf[4];
    bf16_t* Qh[4]; bf16_t* Kh[4]; bf16_t* Vh[4];
    float* xs[4]; float* xc[4];
    bf16_t* xsT[4]; bf16_t* xcT[4];
    float* xsm[4]; float* xcm[4];
    bf16_t* catT[4]; float* h1[4]; bf16_t* hT[4];
    for (int sb = 0; sb < 4; ++sb) {
        descT[sb] = carve<bf16_t>(p, (size_t)Nn * Dd);
        Qf[sb] = carve<float>(p, DN);
        Kf[sb] = carve<float>(p, DN);
        Vf[sb] = carve<float>(p, DN);
        Qh[sb] = carve<bf16_t>(p, (size_t)4 * Nn * 64);
        Kh[sb] = carve<bf16_t>(p, (size_t)4 * Nn * 64);
        Vh[sb] = carve<bf16_t>(p, (size_t)4 * 64 * Nn);
        xs[sb] = carve<float>(p, DN);
        xc[sb] = carve<float>(p, DN);
        xsT[sb] = carve<bf16_t>(p, (size_t)Nn * Dd);
        xcT[sb] = carve<bf16_t>(p, (size_t)Nn * Dd);
        xsm[sb] = carve<float>(p, DN);
        xcm[sb] = carve<float>(p, DN);
        catT[sb] = carve<bf16_t>(p, (size_t)Nn * D3);
        h1[sb]  = carve<float>(p, (size_t)D2 * Nn);
        hT[sb]  = carve<bf16_t>(p, (size_t)Nn * D2);
    }

    auto grid1 = [](size_t n) { return dim3((unsigned)((n + 255) / 256)); };

    // 1) weights -> bf16
    pack_bf16_kernel<<<grid1(Dd*Dd), 256, 0, stream>>>(Wq, Wqb, Dd*Dd);
    pack_bf16_kernel<<<grid1(Dd*Dd), 256, 0, stream>>>(Wk, Wkb, Dd*Dd);
    pack_bf16_kernel<<<grid1(Dd*Dd), 256, 0, stream>>>(Wv, Wvb, Dd*Dd);
    pack_bf16_kernel<<<grid1(Dd*Dd), 256, 0, stream>>>(Wm, Wmb, Dd*Dd);
    pack_bf16_kernel<<<grid1(D2*D3), 256, 0, stream>>>(W1, W1b, D2*D3);
    pack_bf16_kernel<<<grid1(Dd*D2), 256, 0, stream>>>(W2, W2b, Dd*D2);

    // 2) desc^T bf16
    for (int s = 0; s < 2; ++s)
        for (int b = 0; b < 2; ++b) {
            int sb = s * 2 + b;
            packT_kernel<<<grid1(DN), 256, 0, stream>>>(desc[s] + b * DN, descT[sb],
                                                        Dd, Nn, Dd, 0);
        }

    // 3) Q/K/V projections (WMMA GEMMs)
    dim3 gB(128);
    dim3 gGemm256(Nn / 64, Dd / 32);
    for (int sb = 0; sb < 4; ++sb) {
        gemm_kernel<<<gGemm256, gB, 0, stream>>>(Wqb, descT[sb], bq, nullptr, Qf[sb], Dd, Nn, Dd);
        gemm_kernel<<<gGemm256, gB, 0, stream>>>(Wkb, descT[sb], bk, nullptr, Kf[sb], Dd, Nn, Dd);
        gemm_kernel<<<gGemm256, gB, 0, stream>>>(Wvb, descT[sb], bv, nullptr, Vf[sb], Dd, Nn, Dd);
    }

    // 4) per-head repack
    for (int sb = 0; sb < 4; ++sb)
        head_pack_kernel<<<grid1((size_t)4 * 64 * Nn), 256, 0, stream>>>(
            Qf[sb], Kf[sb], Vf[sb], Qh[sb], Kh[sb], Vh[sb]);

    // 5) attention: self (both sides) + cross (both directions)
    dim3 gAttn(Nn / 32, 4);
    for (int b = 0; b < 2; ++b) {
        int s0 = 0 * 2 + b, s1 = 1 * 2 + b;
        attn_kernel<<<gAttn, gB, 0, stream>>>(Qh[s0], Kh[s0], Vh[s0], xs[s0]); // xs0
        attn_kernel<<<gAttn, gB, 0, stream>>>(Qh[s1], Kh[s1], Vh[s1], xs[s1]); // xs1
        attn_kernel<<<gAttn, gB, 0, stream>>>(Qh[s0], Kh[s1], Vh[s1], xc[s0]); // xc0
        attn_kernel<<<gAttn, gB, 0, stream>>>(Kh[s1], Qh[s0], Vh[s0], xc[s1]); // xc1 (S^T path)
    }

    // 6) transpose attention outputs, 7) merge GEMMs
    for (int sb = 0; sb < 4; ++sb) {
        packT_kernel<<<grid1(DN), 256, 0, stream>>>(xs[sb], xsT[sb], Dd, Nn, Dd, 0);
        packT_kernel<<<grid1(DN), 256, 0, stream>>>(xc[sb], xcT[sb], Dd, Nn, Dd, 0);
    }
    for (int sb = 0; sb < 4; ++sb) {
        gemm_kernel<<<gGemm256, gB, 0, stream>>>(Wmb, xsT[sb], bm, nullptr, xsm[sb], Dd, Nn, Dd);
        gemm_kernel<<<gGemm256, gB, 0, stream>>>(Wmb, xcT[sb], bm, nullptr, xcm[sb], Dd, Nn, Dd);
    }

    // 8) cat^T = [desc ; xs_m ; xc_m]^T  (bf16 [N, 768])
    for (int s = 0; s < 2; ++s)
        for (int b = 0; b < 2; ++b) {
            int sb = s * 2 + b;
            packT_kernel<<<grid1(DN), 256, 0, stream>>>(desc[s] + b * DN, catT[sb], Dd, Nn, D3, 0);
            packT_kernel<<<grid1(DN), 256, 0, stream>>>(xsm[sb], catT[sb], Dd, Nn, D3, Dd);
            packT_kernel<<<grid1(DN), 256, 0, stream>>>(xcm[sb], catT[sb], Dd, Nn, D3, 2 * Dd);
        }

    // 9) MLP first GEMM, 10) LN+GELU, 11) MLP second GEMM + residual -> d_out
    dim3 gGemm512(Nn / 64, D2 / 32);
    for (int sb = 0; sb < 4; ++sb)
        gemm_kernel<<<gGemm512, gB, 0, stream>>>(W1b, catT[sb], b1, nullptr, h1[sb], D2, Nn, D3);
    for (int sb = 0; sb < 4; ++sb)
        ln_gelu_kernel<<<dim3(Nn), 256, 0, stream>>>(h1[sb], lng, lnb, hT[sb]);
    for (int s = 0; s < 2; ++s)
        for (int b = 0; b < 2; ++b) {
            int sb = s * 2 + b;
            gemm_kernel<<<gGemm256, gB, 0, stream>>>(
                W2b, hT[sb], b2, desc[s] + b * DN,
                out + (size_t)s * 2 * DN + b * DN, Dd, Nn, D2);
        }
}
